// WedoctorAttention_33200097198843
// MI455X (gfx1250) — matrix operand order repaired and checked by reading, not compiled, not run on hardware
//
#include <hip/hip_runtime.h>
#include <hip/hip_bf16.h>

// ---------------------------------------------------------------------------
// Problem constants (from the reference)
// ---------------------------------------------------------------------------
constexpr int cB   = 2;
constexpr int cS   = 2048;
constexpr int cHID = 2048;
constexpr int cNH  = 32;
constexpr int cNKV = 8;
constexpr int cHD  = 64;          // head dim
constexpr int cM   = cB * cS;     // 4096 rows of X
constexpr int cKVW = cNKV * cHD;  // 512

typedef __attribute__((ext_vector_type(16))) __bf16       v16bf;
typedef __attribute__((ext_vector_type(8)))  float        v8f;
typedef __attribute__((ext_vector_type(4)))  unsigned int u32x4;

union BF16x16 { v16bf v; u32x4 q[2]; };

static __device__ __forceinline__ __bf16 f2bf(float f) {
  union { float f; unsigned int u; } x; x.f = f;
  unsigned int r = x.u + 0x7fffu + ((x.u >> 16) & 1u);   // RNE
  union { unsigned short u; __bf16 b; } y; y.u = (unsigned short)(r >> 16);
  return y.b;
}

// ---------------------------------------------------------------------------
// Async global -> LDS copy (CDNA5 GLOBAL_LOAD_ASYNC_TO_LDS_B128, ASYNCcnt).
// Low 32 bits of a flat shared address are the LDS byte offset (ISA 10.2).
// ---------------------------------------------------------------------------
static __device__ __forceinline__ void async_ld_b128(unsigned lds_off, const void* g) {
  asm volatile("global_load_async_to_lds_b128 %0, %1, off"
               :: "v"(lds_off), "v"(g) : "memory");
}
static __device__ __forceinline__ void wait_async0() {
  asm volatile("s_wait_asynccnt 0x0" ::: "memory");
}

// ---------------------------------------------------------------------------
// fp32 -> bf16 elementwise convert (used for X only)
// ---------------------------------------------------------------------------
__global__ void wd_cvt_bf16(const float* __restrict__ X, __bf16* __restrict__ Y, int n) {
  int i = blockIdx.x * blockDim.x + threadIdx.x;
  if (i < n) Y[i] = f2bf(X[i]);
}

// ---------------------------------------------------------------------------
// Weight transpose + convert:  WT[n*K + k] = (bf16) W[k*N + n]   (done once)
// ---------------------------------------------------------------------------
__global__ void wd_cvt_wT(const float* __restrict__ W, __bf16* __restrict__ WT,
                          int K, int N) {
  int idx = blockIdx.x * blockDim.x + threadIdx.x;
  if (idx >= K * N) return;
  int k = idx % K, n = idx / K;
  WT[(size_t)n * K + k] = f2bf(W[(size_t)k * N + n]);
}

// ---------------------------------------------------------------------------
// V transpose + convert:  VT[(b*KVW + col)*S + s] = (bf16) Vf[(b*S+s)*KVW + col]
// ---------------------------------------------------------------------------
__global__ void wd_cvt_vT(const float* __restrict__ Vf, __bf16* __restrict__ VT) {
  int idx = blockIdx.x * blockDim.x + threadIdx.x;
  if (idx >= cM * cKVW) return;
  int s = idx % cS;
  int t = idx / cS;
  int col = t % cKVW, b = t / cKVW;
  VT[idx] = f2bf(Vf[((size_t)b * cS + s) * cKVW + col]);
}

// ---------------------------------------------------------------------------
// RoPE: fp32 in, bf16 out.  One thread handles one (row, head, i<32) pair.
// ---------------------------------------------------------------------------
__global__ void wd_rope_cvt(const float* __restrict__ X, const int* __restrict__ pos,
                            __bf16* __restrict__ Y, int nheads, int total) {
  int idx = blockIdx.x * blockDim.x + threadIdx.x;
  if (idx >= total) return;
  int i   = idx & 31;
  int h   = (idx >> 5) % nheads;
  int row = idx / (nheads * 32);
  float p = (float)pos[row];
  float inv_freq = __powf(10000.0f, -(float)i * (1.0f / 32.0f));
  float ang = p * inv_freq;
  float c, s;
  __sincosf(ang, &c, &s);
  size_t base = (size_t)row * (nheads * cHD) + (size_t)h * cHD + i;
  float x1 = X[base], x2 = X[base + 32];
  Y[base]      = f2bf(x1 * c - x2 * s);
  Y[base + 32] = f2bf(x2 * c + x1 * s);
}

// ---------------------------------------------------------------------------
// WMMA bf16 GEMM:  C(f32, MxN) = A(bf16 MxK, row-major) * BT(bf16 NxK)^T + bias
// Block tile 128x128, BK=64, double-buffered LDS, async global->LDS staging:
//   wait_asynccnt; barrier; issue async t+1; 16 WMMAs from buf[t&1].
// ---------------------------------------------------------------------------
__global__ __launch_bounds__(256)
void wd_gemm_bf16(const __bf16* __restrict__ A, const __bf16* __restrict__ BT,
                  const float* __restrict__ bias, float* __restrict__ C,
                  int M, int N, int K) {
  __shared__ __align__(16) __bf16 As[2][128 * 64];   // [row][k]
  __shared__ __align__(16) __bf16 Bs[2][128 * 64];   // [n][k]

  const int tid  = threadIdx.x;
  const int lane = tid & 31, wid = tid >> 5;
  const int hf   = lane >> 4, l16 = lane & 15;
  const int bm   = blockIdx.y * 128, bn = blockIdx.x * 128;
  const int wm   = (wid & 3) * 32, wn = (wid >> 2) * 64;

  // staging coords: each thread moves 32 contiguous bf16 of one row per array
  const int r_st = tid >> 1, c_st = (tid & 1) * 32;
  const __bf16* gA = A  + (size_t)(bm + r_st) * K + c_st;
  const __bf16* gB = BT + (size_t)(bn + r_st) * K + c_st;
  const unsigned lA = (unsigned)(size_t)&As[0][r_st * 64 + c_st];
  const unsigned lB = (unsigned)(size_t)&Bs[0][r_st * 64 + c_st];
  const unsigned bufBytes = 128 * 64 * 2;            // 16 KB per buffer

  auto stage = [&](int t) {
    unsigned boff = (t & 1) ? bufBytes : 0u;
    int k0 = t << 6;
#pragma unroll
    for (int j = 0; j < 4; ++j) {
      async_ld_b128(lA + boff + j * 16, gA + k0 + j * 8);
      async_ld_b128(lB + boff + j * 16, gB + k0 + j * 8);
    }
  };

  v8f acc[2][4] = {};
  const int T = K >> 6;
  stage(0);
  for (int t = 0; t < T; ++t) {
    wait_async0();            // tile t landed in LDS (this wave's portion)
    __syncthreads();          // everyone's portion landed
    if (t + 1 < T) stage(t + 1);                 // DMA next tile over compute
    if (t + 2 < T) {                             // deepen: prefetch t+2 to L2
      __builtin_prefetch(gA + ((t + 2) << 6), 0, 3);
      __builtin_prefetch(gB + ((t + 2) << 6), 0, 3);
    }
    const int buf = t & 1;

#pragma unroll
    for (int kk = 0; kk < 64; kk += 32) {
      BF16x16 af[2], bf[4];
#pragma unroll
      for (int mt = 0; mt < 2; ++mt) {
        const __bf16* p = &As[buf][(wm + mt * 16 + l16) * 64 + kk + hf * 8];
        af[mt].q[0] = *(const u32x4*)(p);
        af[mt].q[1] = *(const u32x4*)(p + 16);
      }
#pragma unroll
      for (int nt = 0; nt < 4; ++nt) {
        const __bf16* p = &Bs[buf][(wn + nt * 16 + l16) * 64 + kk + hf * 8];
        bf[nt].q[0] = *(const u32x4*)(p);
        bf[nt].q[1] = *(const u32x4*)(p + 16);
      }
#pragma unroll
      for (int mt = 0; mt < 2; ++mt)
#pragma unroll
        for (int nt = 0; nt < 4; ++nt)
          acc[mt][nt] = __builtin_amdgcn_wmma_f32_16x16x32_bf16(
              false, af[mt].v, false, bf[nt].v, (short)0, acc[mt][nt],
              false, false);
    }
  }

#pragma unroll
  for (int mt = 0; mt < 2; ++mt)
#pragma unroll
    for (int nt = 0; nt < 4; ++nt) {
      int col = bn + wn + nt * 16 + l16;
      float bias_v = bias ? bias[col] : 0.0f;
#pragma unroll
      for (int r = 0; r < 8; ++r) {
        int row = bm + wm + mt * 16 + r + hf * 8;
        C[(size_t)row * N + col] = acc[mt][nt][r] + bias_v;
      }
    }
}

// ---------------------------------------------------------------------------
// Flash attention (causal, GQA 4:1).  Block = (b, h, 128 q rows), 8 waves.
// K (row-major) and V (pre-transposed [d][S]) staged in double-buffered
// 64-key LDS tiles via async global->LDS DMA + L2 prefetch of tile t+2.
// ---------------------------------------------------------------------------
__global__ __launch_bounds__(256)
void wd_flash_attn(const __bf16* __restrict__ Q, const __bf16* __restrict__ Kk,
                   const __bf16* __restrict__ VT, __bf16* __restrict__ O) {
  __shared__ __align__(16) __bf16 Ks[2][64 * 64];     // [key][d]
  __shared__ __align__(16) __bf16 Vt[2][64 * 64];     // [d][key]
  __shared__ __align__(16) __bf16 Pw[8][16 * 64];     // per-wave P tile [m][key]

  const int tid  = threadIdx.x;
  const int lane = tid & 31, wid = tid >> 5;
  const int hf   = lane >> 4, l16 = lane & 15;
  const int h    = blockIdx.y, hkv = h >> 2, b = blockIdx.z;
  const int q0   = blockIdx.x * 128;
  const int qrow = q0 + wid * 16;                       // wave's first q row
  const int klast = q0 + 127;                           // last key any wave needs

  // Q fragments for k-chunks [0,32) and [32,64) -- live in registers all kernel
  BF16x16 qf[2];
  {
    const __bf16* qp =
        Q + ((size_t)b * cS + qrow + l16) * (cNH * cHD) + (size_t)h * cHD;
#pragma unroll
    for (int c = 0; c < 2; ++c) {
      qf[c].q[0] = *(const u32x4*)(qp + c * 32 + hf * 8);
      qf[c].q[1] = *(const u32x4*)(qp + c * 32 + 16 + hf * 8);
    }
  }

  v8f  o[4] = {};
  float mrow[8], lrow[8];
#pragma unroll
  for (int r = 0; r < 8; ++r) { mrow[r] = -__builtin_inff(); lrow[r] = 0.0f; }
  const float scale = 0.125f;   // 1/sqrt(64)

  // per-thread staging addresses (key/d pattern shared by K and V movers)
  const int skey = tid >> 2, sc16 = (tid & 3) * 16;
  const __bf16* gK = Kk + ((size_t)b * cS + skey) * cKVW + (size_t)hkv * cHD + sc16;
  const __bf16* gV = VT + ((size_t)b * cKVW + hkv * cHD + skey) * cS + sc16;
  const unsigned lK = (unsigned)(size_t)&Ks[0][skey * 64 + sc16];
  const unsigned lV = (unsigned)(size_t)&Vt[0][skey * 64 + sc16];
  const unsigned fb = 64 * 64 * 2;                     // 8 KB per buffer

  auto stage = [&](int kb, int bi) {
    unsigned boff = bi ? fb : 0u;
    const __bf16* sk = gK + (size_t)kb * cKVW;
    const __bf16* sv = gV + kb;
    async_ld_b128(lK + boff,      sk);
    async_ld_b128(lK + boff + 16, sk + 8);
    async_ld_b128(lV + boff,      sv);
    async_ld_b128(lV + boff + 16, sv + 8);
  };

  const int nTiles = (klast >> 6) + 1;
  stage(0, 0);
  for (int ti = 0; ti < nTiles; ++ti) {
    const int kb = ti << 6;
    wait_async0();
    __syncthreads();
    if (ti + 1 < nTiles) stage(kb + 64, (ti + 1) & 1);
    if (ti + 2 < nTiles) {
      __builtin_prefetch(gK + (size_t)(kb + 128) * cKVW, 0, 3);
      __builtin_prefetch(gV + kb + 128, 0, 3);
    }
    if (kb > qrow + 15) continue;   // wave-uniform: fully masked for this wave
    const __bf16* KsC = &Ks[ti & 1][0];
    const __bf16* VtC = &Vt[ti & 1][0];

    // ---- scores S = Q * K^T  (16x64, 4 n-tiles x 2 k-steps) ----
    v8f sacc[4] = {};
#pragma unroll
    for (int nt = 0; nt < 4; ++nt)
#pragma unroll
      for (int c = 0; c < 2; ++c) {
        BF16x16 bfk;
        const __bf16* p = &KsC[(nt * 16 + l16) * 64 + c * 32 + hf * 8];
        bfk.q[0] = *(const u32x4*)(p);
        bfk.q[1] = *(const u32x4*)(p + 16);
        sacc[nt] = __builtin_amdgcn_wmma_f32_16x16x32_bf16(
            false, qf[c].v, false, bfk.v, (short)0, sacc[nt], false, false);
      }

    // ---- scale + causal mask + row max (reduce over 16-lane halves) ----
    float nm[8];
#pragma unroll
    for (int r = 0; r < 8; ++r) {
      int qg = qrow + r + hf * 8;
      float mx = -__builtin_inff();
#pragma unroll
      for (int nt = 0; nt < 4; ++nt) {
        int kg = kb + nt * 16 + l16;
        float s = sacc[nt][r] * scale;
        s = (kg <= qg) ? s : -__builtin_inff();
        sacc[nt][r] = s;
        mx = fmaxf(mx, s);
      }
#pragma unroll
      for (int off = 1; off < 16; off <<= 1)
        mx = fmaxf(mx, __shfl_xor(mx, off, 32));
      nm[r] = fmaxf(mrow[r], mx);
    }

    // ---- P = exp(S - m), running sums, rescale O ----
#pragma unroll
    for (int r = 0; r < 8; ++r) {
      float alpha = __expf(mrow[r] - nm[r]);
      float rs = 0.0f;
#pragma unroll
      for (int nt = 0; nt < 4; ++nt) {
        float p = __expf(sacc[nt][r] - nm[r]);
        sacc[nt][r] = p;
        rs += p;
      }
#pragma unroll
      for (int off = 1; off < 16; off <<= 1)
        rs += __shfl_xor(rs, off, 32);
      lrow[r] = lrow[r] * alpha + rs;
      mrow[r] = nm[r];
#pragma unroll
      for (int nt = 0; nt < 4; ++nt) o[nt][r] *= alpha;
    }

    // ---- D-layout -> A-layout via per-wave LDS round trip ----
#pragma unroll
    for (int r = 0; r < 8; ++r) {
      int m = r + hf * 8;
#pragma unroll
      for (int nt = 0; nt < 4; ++nt)
        Pw[wid][m * 64 + nt * 16 + l16] = f2bf(sacc[nt][r]);
    }
    asm volatile("s_wait_dscnt 0" ::: "memory");   // same-wave LDS RAW

    // ---- O += P * V  (A = P 16x64, B = V tile 32x16 per step) ----
#pragma unroll
    for (int c = 0; c < 2; ++c) {
      BF16x16 pf;
      const __bf16* pp = &Pw[wid][l16 * 64 + c * 32 + hf * 8];
      pf.q[0] = *(const u32x4*)(pp);
      pf.q[1] = *(const u32x4*)(pp + 16);
#pragma unroll
      for (int nt = 0; nt < 4; ++nt) {
        BF16x16 vf;
        const __bf16* vp = &VtC[(nt * 16 + l16) * 64 + c * 32 + hf * 8];
        vf.q[0] = *(const u32x4*)(vp);
        vf.q[1] = *(const u32x4*)(vp + 16);
        o[nt] = __builtin_amdgcn_wmma_f32_16x16x32_bf16(
            false, pf.v, false, vf.v, (short)0, o[nt], false, false);
      }
    }
  }

  // ---- normalize + store bf16 for the Wo GEMM ----
#pragma unroll
  for (int nt = 0; nt < 4; ++nt)
#pragma unroll
    for (int r = 0; r < 8; ++r) {
      float inv = 1.0f / lrow[r];
      size_t row = (size_t)b * cS + qrow + r + hf * 8;
      O[row * (cNH * cHD) + (size_t)h * cHD + nt * 16 + l16] =
          f2bf(o[nt][r] * inv);
    }
}

// ---------------------------------------------------------------------------
// Host-side orchestration
// ---------------------------------------------------------------------------
extern "C" void kernel_launch(void* const* d_in, const int* in_sizes, int n_in,
                              void* d_out, int out_size, void* d_ws, size_t ws_size,
                              hipStream_t stream) {
  const float* hs = (const float*)d_in[0];
  const float* Wq = (const float*)d_in[1];
  const float* bq = (const float*)d_in[2];
  const float* Wk = (const float*)d_in[3];
  const float* bk = (const float*)d_in[4];
  const float* Wv = (const float*)d_in[5];
  const float* bv = (const float*)d_in[6];
  const float* Wo = (const float*)d_in[7];
  const int*   pos = (const int*)d_in[8];
  float* out = (float*)d_out;

  char* ws = (char*)d_ws;
  size_t off = 0;
  auto carve = [&](size_t bytes) { size_t o = off; off += (bytes + 255) & ~(size_t)255; return o; };

  __bf16* Xb  = (__bf16*)(ws + carve((size_t)cM * cHID * 2));
  __bf16* WqT = (__bf16*)(ws + carve((size_t)cHID * cHID * 2));
  __bf16* WkT = (__bf16*)(ws + carve((size_t)cHID * cKVW * 2));
  __bf16* WvT = (__bf16*)(ws + carve((size_t)cHID * cKVW * 2));
  __bf16* WoT = (__bf16*)(ws + carve((size_t)cHID * cHID * 2));
  float*  Qf  = (float*) (ws + carve((size_t)cM * cHID * 4));
  float*  Kf  = (float*) (ws + carve((size_t)cM * cKVW * 4));
  float*  Vf  = (float*) (ws + carve((size_t)cM * cKVW * 4));
  __bf16* Qb  = (__bf16*)(ws + carve((size_t)cM * cHID * 2));
  __bf16* Kb  = (__bf16*)(ws + carve((size_t)cM * cKVW * 2));
  __bf16* VT  = (__bf16*)(ws + carve((size_t)cM * cKVW * 2));
  __bf16* Ob  = (__bf16*)(ws + carve((size_t)cM * cHID * 2));

  // one-time converts / transposes
  wd_cvt_bf16<<<(cM * cHID + 255) / 256, 256, 0, stream>>>(hs, Xb, cM * cHID);
  wd_cvt_wT<<<(cHID * cHID + 255) / 256, 256, 0, stream>>>(Wq, WqT, cHID, cHID);
  wd_cvt_wT<<<(cHID * cKVW + 255) / 256, 256, 0, stream>>>(Wk, WkT, cHID, cKVW);
  wd_cvt_wT<<<(cHID * cKVW + 255) / 256, 256, 0, stream>>>(Wv, WvT, cHID, cKVW);
  wd_cvt_wT<<<(cHID * cHID + 255) / 256, 256, 0, stream>>>(Wo, WoT, cHID, cHID);

  // QKV projections (bf16 WMMA GEMM, fp32 out + bias)
  wd_gemm_bf16<<<dim3(cHID / 128, cM / 128), 256, 0, stream>>>(Xb, WqT, bq, Qf, cM, cHID, cHID);
  wd_gemm_bf16<<<dim3(cKVW / 128, cM / 128), 256, 0, stream>>>(Xb, WkT, bk, Kf, cM, cKVW, cHID);
  wd_gemm_bf16<<<dim3(cKVW / 128, cM / 128), 256, 0, stream>>>(Xb, WvT, bv, Vf, cM, cKVW, cHID);

  // RoPE (Q,K) + downcast; V downcast + transpose to [b][hkv][d][S]
  {
    int totQ = cM * cNH * 32;
    wd_rope_cvt<<<(totQ + 255) / 256, 256, 0, stream>>>(Qf, pos, Qb, cNH, totQ);
    int totK = cM * cNKV * 32;
    wd_rope_cvt<<<(totK + 255) / 256, 256, 0, stream>>>(Kf, pos, Kb, cNKV, totK);
    wd_cvt_vT<<<(cM * cKVW + 255) / 256, 256, 0, stream>>>(Vf, VT);
  }

  // Causal flash attention with GQA sharing (h -> h/4)
  wd_flash_attn<<<dim3(cS / 128, cNH, cB), 256, 0, stream>>>(Qb, Kb, VT, Ob);

  // Output projection: out = Ob @ Wo  (no bias)
  wd_gemm_bf16<<<dim3(cHID / 128, cM / 128), 256, 0, stream>>>(Ob, WoT, nullptr, out, cM, cHID, cHID);
}
